// KimiAudioMIMODecoderLayer_73426760893150
// MI455X (gfx1250) — matrix operand order repaired and checked by reading, not compiled, not run on hardware
//
#include <hip/hip_runtime.h>
#include <hip/hip_bf16.h>

// ---------------------------------------------------------------------------
// Qwen2-style decoder layer on MI455X (gfx1250, wave32).
// Strategy: one bf16 conversion pass over weights/activations (~40us of the
// 23.3 TB/s HBM), then all GEMMs are pure-bf16 v_wmma_f32_16x16x32_bf16 with
// double-buffered global->LDS staging via global_load_async_to_lds_b128
// (ASYNCcnt pipeline per ISA ch.10), 8 WMMAs per wave per K-tile.
// ---------------------------------------------------------------------------

typedef __attribute__((ext_vector_type(16))) __bf16 v16bf;
typedef __attribute__((ext_vector_type(8)))  __bf16 v8bf;
typedef __attribute__((ext_vector_type(8)))  float  v8f;
typedef __attribute__((ext_vector_type(4)))  int    v4i;
typedef __bf16 bf16_t;

#define S_LEN 2048
#define HDIM  3584
#define NHEAD 28
#define NKVH  4
#define HDHD  128
#define IDIM  18944
#define GQA   7
#define MLP_CHUNK 512

#define AS1 __attribute__((address_space(1)))
#define AS3 __attribute__((address_space(3)))

#if __has_builtin(__builtin_amdgcn_global_load_async_to_lds_b128)
#define HAVE_ASYNC 1
#else
#define HAVE_ASYNC 0
#endif

__device__ __forceinline__ bf16_t to_bf16(float f) {
    union { float f; unsigned u; } v; v.f = f;
    unsigned r = (v.u + 0x7FFFu + ((v.u >> 16) & 1u)) >> 16;   // RNE
    union { unsigned short s; bf16_t b; } o; o.s = (unsigned short)r;
    return o.b;
}
__device__ __forceinline__ unsigned short to_bf16_bits(float f) {
    union { float f; unsigned u; } v; v.f = f;
    return (unsigned short)((v.u + 0x7FFFu + ((v.u >> 16) & 1u)) >> 16);
}
__device__ __forceinline__ float bf16_to_f32(bf16_t b) {
    union { unsigned short s; bf16_t b; } i; i.b = b;
    union { unsigned u; float f; } o; o.u = ((unsigned)i.s) << 16;
    return o.f;
}

__device__ __forceinline__ void wait_async_copies() {
#if HAVE_ASYNC
#if __has_builtin(__builtin_amdgcn_s_wait_asynccnt)
    __builtin_amdgcn_s_wait_asynccnt(0);
#else
    asm volatile("s_wait_asynccnt 0x0" ::: "memory");
#endif
#endif
}

// ---------------------------------------------------------------------------
// Pure-bf16 tiled WMMA GEMM:  C[m,n] = sum_k A[m,k]*B[n,k] (+bias[n])(+add[m,n])
// A: (M,K) K-contiguous bf16;  B: (N,K) K-contiguous bf16;  C: fp32.
// Block tile 128x128x32, 256 threads = 8 waves (2x4), wave tile 64x32 = 4x2
// accumulators. Double-buffered LDS, async global->LDS copies.
// ---------------------------------------------------------------------------
#define BM 128
#define BN 128
#define BK 32

// Copy one 128x32 bf16 tile (8KB) into LDS: 2 shots x 256 lanes x 16B.
__device__ __forceinline__ void stage_tile(const bf16_t* __restrict__ g, long ld,
                                           long row0, int k0,
                                           bf16_t* __restrict__ lds, int tid)
{
    #pragma unroll
    for (int sh = 0; sh < 2; ++sh) {
        int e = (sh * 256 + tid) * 8;         // element index in tile
        int r = e >> 5;                       // /BK
        int c = e & 31;
        const bf16_t* gp = g + (row0 + r) * ld + (k0 + c);
        bf16_t* lp = lds + r * BK + c;
#if HAVE_ASYNC
        __builtin_amdgcn_global_load_async_to_lds_b128(
            (AS1 v4i*)gp, (AS3 v4i*)lp, 0, 0);
#else
        __builtin_prefetch(gp, 0, 1);
        *(uint4*)lp = *(const uint4*)gp;
#endif
    }
}

__global__ __launch_bounds__(256) void gemm_bf16_wmma_kernel(
    const bf16_t* __restrict__ A, int lda,
    const bf16_t* __restrict__ B, int ldb,
    float* __restrict__ C, int ldc,
    const float* __restrict__ bias,
    const float* __restrict__ addsrc, int ldadd,
    int M, int N, int K)
{
    __shared__ alignas(16) bf16_t As[2][BM * BK];   // 2 x 8KB
    __shared__ alignas(16) bf16_t Bs[2][BN * BK];   // 2 x 8KB

    const int tid   = threadIdx.x;
    const int lane  = tid & 31;
    const int wave  = tid >> 5;        // 0..7
    const int waveM = wave >> 2;       // 0..1 -> 64-row band
    const int waveN = wave & 3;        // 0..3 -> 32-col band
    const int l15   = lane & 15;
    const int hi    = lane >> 4;       // lane[4]: K-half select (contiguous k = 16*hi + s)

    const long blockM = (long)blockIdx.y * BM;
    const long blockN = (long)blockIdx.x * BN;

    v8f acc[4][2] = {};

    const int nk = K / BK;

    // prologue: stage tile 0 into buffer 0
    stage_tile(A, lda, blockM, 0, As[0], tid);
    stage_tile(B, ldb, blockN, 0, Bs[0], tid);

    for (int kt = 0; kt < nk; ++kt) {
        wait_async_copies();     // current tile's copies (this wave) done
        __syncthreads();         // whole WGP's copies / previous compute done

        if (kt + 1 < nk) {       // overlap: issue next tile while computing
            stage_tile(A, lda, blockM, (kt + 1) * BK, As[(kt + 1) & 1], tid);
            stage_tile(B, ldb, blockN, (kt + 1) * BK, Bs[(kt + 1) & 1], tid);
        }

        const bf16_t* as = As[kt & 1];
        const bf16_t* bs = Bs[kt & 1];

        // A fragments (16x32): lane = M row, K halves selected by lane[4]
        v16bf af[4];
        #pragma unroll
        for (int mi = 0; mi < 4; ++mi) {
            int mrow = waveM * 64 + mi * 16 + l15;
            v8bf lo  = *(const v8bf*)&as[mrow * BK + hi * 16];
            v8bf hi8 = *(const v8bf*)&as[mrow * BK + hi * 16 + 8];
            af[mi] = __builtin_shufflevector(lo, hi8,
                0,1,2,3,4,5,6,7,8,9,10,11,12,13,14,15);
        }
        // B fragments (32x16): lane = N col, K half by lane[4]
        v16bf bfrag[2];
        #pragma unroll
        for (int ni = 0; ni < 2; ++ni) {
            int ncol = waveN * 32 + ni * 16 + l15;
            bfrag[ni] = *(const v16bf*)&bs[ncol * BK + hi * 16];
        }
        // 8 back-to-back WMMAs with operand reuse
        #pragma unroll
        for (int mi = 0; mi < 4; ++mi)
            #pragma unroll
            for (int ni = 0; ni < 2; ++ni)
                acc[mi][ni] = __builtin_amdgcn_wmma_f32_16x16x32_bf16(
                    false, af[mi], false, bfrag[ni], (short)0, acc[mi][ni],
                    false, false);
    }

    // store: lane = N col (l15), VGPR r = M row (+8 for lanes 16..31)
    #pragma unroll
    for (int mi = 0; mi < 4; ++mi) {
        #pragma unroll
        for (int ni = 0; ni < 2; ++ni) {
            long nc = blockN + waveN * 32 + ni * 16 + l15;
            #pragma unroll
            for (int r = 0; r < 8; ++r) {
                long mr = blockM + waveM * 64 + mi * 16 + r + hi * 8;
                float v = acc[mi][ni][r];
                if (bias)   v += bias[nc];
                if (addsrc) v += addsrc[mr * (long)ldadd + nc];
                C[mr * (long)ldc + nc] = v;
            }
        }
    }
}

// ---------------------------------------------------------------------------
// fp32 -> bf16 conversion, 4 elements/thread, packed 8B stores.
// ---------------------------------------------------------------------------
__global__ __launch_bounds__(256) void cvt_f32_bf16_kernel(
    const float* __restrict__ src, bf16_t* __restrict__ dst, long n)
{
    long i4 = blockIdx.x * 256L + threadIdx.x;
    if (i4 * 4 >= n) return;
    float4 f = ((const float4*)src)[i4];
    uint2 p;
    p.x = (unsigned)to_bf16_bits(f.x) | ((unsigned)to_bf16_bits(f.y) << 16);
    p.y = (unsigned)to_bf16_bits(f.z) | ((unsigned)to_bf16_bits(f.w) << 16);
    ((uint2*)dst)[i4] = p;
}

// ---------------------------------------------------------------------------
// RMSNorm: fp32 in -> bf16 out, one block per row.
// ---------------------------------------------------------------------------
__global__ __launch_bounds__(256) void rmsnorm_bf16_kernel(
    const float* __restrict__ x, const float* __restrict__ w,
    bf16_t* __restrict__ out, int H)
{
    const long row = blockIdx.x;
    const float* xr = x + row * (long)H;
    bf16_t* orow = out + row * (long)H;

    float ss = 0.f;
    for (int i = threadIdx.x; i < H; i += 256) { float v = xr[i]; ss += v * v; }
    __shared__ float red[256];
    red[threadIdx.x] = ss;
    __syncthreads();
    for (int s = 128; s > 0; s >>= 1) {
        if (threadIdx.x < s) red[threadIdx.x] += red[threadIdx.x + s];
        __syncthreads();
    }
    float inv = rsqrtf(red[0] / (float)H + 1e-6f);
    for (int i = threadIdx.x; i < H; i += 256)
        orow[i] = to_bf16(xr[i] * inv * w[i]);
}

// ---------------------------------------------------------------------------
// RoPE: fp32 in -> bf16 out. One thread per (s, head, hd<64) rotation pair.
// ---------------------------------------------------------------------------
__global__ __launch_bounds__(256) void rope_cvt_kernel(
    const float* __restrict__ qk, bf16_t* __restrict__ out,
    const int* __restrict__ pos_ids, int nheads, int S)
{
    long idx = blockIdx.x * 256L + threadIdx.x;
    long total = (long)S * nheads * 64;
    if (idx >= total) return;
    int hd = (int)(idx % 64);
    long sh = idx / 64;
    int h = (int)(sh % nheads);
    long s = sh / nheads;

    float pos = (float)pos_ids[s];
    float invf = __powf(1.0e6f, -(float)hd * (1.0f / 64.0f));
    float ang = pos * invf;
    float c = __cosf(ang), sn = __sinf(ang);

    const float* bi = qk + (s * nheads + h) * 128;
    bf16_t* bo = out + (s * nheads + h) * 128;
    float x1 = bi[hd], x2 = bi[hd + 64];
    bo[hd]      = to_bf16(x1 * c - x2 * sn);
    bo[hd + 64] = to_bf16(x2 * c + x1 * sn);
}

// ---------------------------------------------------------------------------
// V transpose + convert: vT[d, t] = (bf16) v[t, d].  v: (S, 512), vT: (512, S)
// ---------------------------------------------------------------------------
__global__ __launch_bounds__(256) void vtrans_kernel(
    const float* __restrict__ v, bf16_t* __restrict__ vT, int S, int D)
{
    long idx = blockIdx.x * 256L + threadIdx.x;
    if (idx >= (long)S * D) return;
    int t = (int)(idx % S);
    int d = (int)(idx / S);
    vT[(long)d * S + t] = to_bf16(v[(long)t * D + d]);
}

// ---------------------------------------------------------------------------
// Row softmax with pre-scale, fp32 in -> bf16 out.
// ---------------------------------------------------------------------------
__global__ __launch_bounds__(256) void softmax_bf16_kernel(
    const float* __restrict__ scores, bf16_t* __restrict__ probs,
    int cols, float scale)
{
    const long row = blockIdx.x;
    const float* sr = scores + row * (long)cols;
    bf16_t* pr = probs + row * (long)cols;
    __shared__ float red[256];

    float mx = -3.0e38f;
    for (int i = threadIdx.x; i < cols; i += 256)
        mx = fmaxf(mx, sr[i] * scale);
    red[threadIdx.x] = mx;
    __syncthreads();
    for (int s = 128; s > 0; s >>= 1) {
        if (threadIdx.x < s) red[threadIdx.x] = fmaxf(red[threadIdx.x], red[threadIdx.x + s]);
        __syncthreads();
    }
    float m = red[0];
    __syncthreads();

    float sum = 0.f;
    for (int i = threadIdx.x; i < cols; i += 256)
        sum += __expf(sr[i] * scale - m);
    red[threadIdx.x] = sum;
    __syncthreads();
    for (int s = 128; s > 0; s >>= 1) {
        if (threadIdx.x < s) red[threadIdx.x] += red[threadIdx.x + s];
        __syncthreads();
    }
    float inv = 1.0f / red[0];
    for (int i = threadIdx.x; i < cols; i += 256)
        pr[i] = to_bf16(__expf(sr[i] * scale - m) * inv);
}

// ---------------------------------------------------------------------------
// SwiGLU: gu = (bf16)( silu(gate) * up ), fp32 inputs.
// ---------------------------------------------------------------------------
__global__ __launch_bounds__(256) void silu_mul_bf16_kernel(
    const float* __restrict__ gate, const float* __restrict__ up,
    bf16_t* __restrict__ gu, long n)
{
    long i = blockIdx.x * 256L + threadIdx.x;
    if (i >= n) return;
    float g = gate[i];
    float sg = g / (1.0f + __expf(-g));
    gu[i] = to_bf16(sg * up[i]);
}

// ---------------------------------------------------------------------------
extern "C" void kernel_launch(void* const* d_in, const int* in_sizes, int n_in,
                              void* d_out, int out_size, void* d_ws, size_t ws_size,
                              hipStream_t stream)
{
    (void)in_sizes; (void)n_in; (void)out_size; (void)ws_size;

    const float* hs      = (const float*)d_in[0];
    const int*   pos_ids = (const int*)  d_in[1];
    const float* q_w     = (const float*)d_in[2];
    const float* q_b     = (const float*)d_in[3];
    const float* k_w     = (const float*)d_in[4];
    const float* k_b     = (const float*)d_in[5];
    const float* v_w     = (const float*)d_in[6];
    const float* v_b     = (const float*)d_in[7];
    const float* o_w     = (const float*)d_in[8];
    const float* gate_w  = (const float*)d_in[9];
    const float* up_w    = (const float*)d_in[10];
    const float* down_w  = (const float*)d_in[11];
    const float* ln1_w   = (const float*)d_in[12];
    const float* ln2_w   = (const float*)d_in[13];
    float* out = (float*)d_out;

    const int QD = NHEAD * HDHD;   // 3584
    const int KD = NKVH * HDHD;    // 512

    // ---- workspace carve (byte cursor, 16B aligned chunks) ----
    char* cur = (char*)d_ws;
    auto alloc_f = [&](long n) { float* p = (float*)cur;  cur += n * 4;              return p; };
    auto alloc_b = [&](long n) { bf16_t* p = (bf16_t*)cur; cur += ((n * 2 + 15) & ~15L); return p; };

    bf16_t* qwb = alloc_b((long)QD * HDIM);
    bf16_t* kwb = alloc_b((long)KD * HDIM);
    bf16_t* vwb = alloc_b((long)KD * HDIM);
    bf16_t* owb = alloc_b((long)HDIM * HDIM);
    bf16_t* gwb = alloc_b((long)IDIM * HDIM);
    bf16_t* uwb = alloc_b((long)IDIM * HDIM);
    bf16_t* dwb = alloc_b((long)HDIM * IDIM);

    bf16_t* x1b    = alloc_b((long)S_LEN * HDIM);
    float*  q      = alloc_f((long)S_LEN * QD);
    float*  k      = alloc_f((long)S_LEN * KD);
    float*  v      = alloc_f((long)S_LEN * KD);
    bf16_t* qb     = alloc_b((long)S_LEN * QD);
    bf16_t* kb     = alloc_b((long)S_LEN * KD);
    bf16_t* vTb    = alloc_b((long)KD * S_LEN);
    float*  scores = alloc_f((long)S_LEN * S_LEN);
    bf16_t* probsb = alloc_b((long)S_LEN * S_LEN);
    float*  attn   = alloc_f((long)S_LEN * HDIM);
    bf16_t* attnb  = alloc_b((long)S_LEN * HDIM);
    float*  hidden = alloc_f((long)S_LEN * HDIM);
    bf16_t* x2b    = alloc_b((long)S_LEN * HDIM);
    float*  gbuf   = alloc_f((long)MLP_CHUNK * IDIM);
    float*  ubuf   = alloc_f((long)MLP_CHUNK * IDIM);
    bf16_t* gub    = alloc_b((long)MLP_CHUNK * IDIM);

    auto cvt = [&](const float* s, bf16_t* d, long n) {
        cvt_f32_bf16_kernel<<<(unsigned)((n / 4 + 255) / 256), 256, 0, stream>>>(s, d, n);
    };

    // ---- 0) weights -> bf16 (single streaming pass) ----
    cvt(q_w,    qwb, (long)QD * HDIM);
    cvt(k_w,    kwb, (long)KD * HDIM);
    cvt(v_w,    vwb, (long)KD * HDIM);
    cvt(o_w,    owb, (long)HDIM * HDIM);
    cvt(gate_w, gwb, (long)IDIM * HDIM);
    cvt(up_w,   uwb, (long)IDIM * HDIM);
    cvt(down_w, dwb, (long)HDIM * IDIM);

    // ---- 1) ln1 -> bf16 ----
    rmsnorm_bf16_kernel<<<S_LEN, 256, 0, stream>>>(hs, ln1_w, x1b, HDIM);

    // ---- 2) QKV projections ----
    gemm_bf16_wmma_kernel<<<dim3(QD / BN, S_LEN / BM), 256, 0, stream>>>(
        x1b, HDIM, qwb, HDIM, q, QD, q_b, nullptr, 0, S_LEN, QD, HDIM);
    gemm_bf16_wmma_kernel<<<dim3(KD / BN, S_LEN / BM), 256, 0, stream>>>(
        x1b, HDIM, kwb, HDIM, k, KD, k_b, nullptr, 0, S_LEN, KD, HDIM);
    gemm_bf16_wmma_kernel<<<dim3(KD / BN, S_LEN / BM), 256, 0, stream>>>(
        x1b, HDIM, vwb, HDIM, v, KD, v_b, nullptr, 0, S_LEN, KD, HDIM);

    // ---- 3) RoPE -> bf16 q/k; V -> bf16 transpose ----
    {
        long tq = (long)S_LEN * NHEAD * 64;
        rope_cvt_kernel<<<(unsigned)((tq + 255) / 256), 256, 0, stream>>>(q, qb, pos_ids, NHEAD, S_LEN);
        long tk = (long)S_LEN * NKVH * 64;
        rope_cvt_kernel<<<(unsigned)((tk + 255) / 256), 256, 0, stream>>>(k, kb, pos_ids, NKVH, S_LEN);
        long tv = (long)S_LEN * KD;
        vtrans_kernel<<<(unsigned)((tv + 255) / 256), 256, 0, stream>>>(v, vTb, S_LEN, KD);
    }

    // ---- 4) attention per head (GQA kv = h/7) ----
    const float scale = 0.088388347648318447f;  // 1/sqrt(128)
    for (int h = 0; h < NHEAD; ++h) {
        int kv = h / GQA;
        gemm_bf16_wmma_kernel<<<dim3(S_LEN / BN, S_LEN / BM), 256, 0, stream>>>(
            qb + (long)h * HDHD, QD, kb + (long)kv * HDHD, KD,
            scores, S_LEN, nullptr, nullptr, 0, S_LEN, S_LEN, HDHD);
        softmax_bf16_kernel<<<S_LEN, 256, 0, stream>>>(scores, probsb, S_LEN, scale);
        gemm_bf16_wmma_kernel<<<dim3(HDHD / BN, S_LEN / BM), 256, 0, stream>>>(
            probsb, S_LEN, vTb + (long)kv * HDHD * S_LEN, S_LEN,
            attn + (long)h * HDHD, HDIM, nullptr, nullptr, 0, S_LEN, HDHD, S_LEN);
    }
    cvt(attn, attnb, (long)S_LEN * HDIM);

    // ---- 5) O projection + residual ----
    gemm_bf16_wmma_kernel<<<dim3(HDIM / BN, S_LEN / BM), 256, 0, stream>>>(
        attnb, HDIM, owb, HDIM, hidden, HDIM, nullptr, hs, HDIM,
        S_LEN, HDIM, HDIM);

    // ---- 6) ln2 -> bf16 ----
    rmsnorm_bf16_kernel<<<S_LEN, 256, 0, stream>>>(hidden, ln2_w, x2b, HDIM);

    // ---- 7) MLP in 4 row-chunks of 512 ----
    for (int c = 0; c < S_LEN / MLP_CHUNK; ++c) {
        const bf16_t* xc = x2b + (long)c * MLP_CHUNK * HDIM;
        gemm_bf16_wmma_kernel<<<dim3(IDIM / BN, MLP_CHUNK / BM), 256, 0, stream>>>(
            xc, HDIM, gwb, HDIM, gbuf, IDIM, nullptr, nullptr, 0,
            MLP_CHUNK, IDIM, HDIM);
        gemm_bf16_wmma_kernel<<<dim3(IDIM / BN, MLP_CHUNK / BM), 256, 0, stream>>>(
            xc, HDIM, uwb, HDIM, ubuf, IDIM, nullptr, nullptr, 0,
            MLP_CHUNK, IDIM, HDIM);
        long n = (long)MLP_CHUNK * IDIM;
        silu_mul_bf16_kernel<<<(unsigned)((n + 255) / 256), 256, 0, stream>>>(gbuf, ubuf, gub, n);
        gemm_bf16_wmma_kernel<<<dim3(HDIM / BN, MLP_CHUNK / BM), 256, 0, stream>>>(
            gub, IDIM, dwb, IDIM, out + (long)c * MLP_CHUNK * HDIM, HDIM,
            nullptr, hidden + (long)c * MLP_CHUNK * HDIM, HDIM,
            MLP_CHUNK, HDIM, IDIM);
    }
}
